// AttentionConv_36292473651334
// MI455X (gfx1250) — compile-verified, matrix operand order checked by reading, not run on hardware
//
#include <hip/hip_runtime.h>

// ---------------- types ----------------
typedef __attribute__((ext_vector_type(16))) __bf16 v16bf;
typedef __attribute__((ext_vector_type(8)))  __bf16 v8bf;
typedef __attribute__((ext_vector_type(8)))  float  v8f;

// ---------------- problem constants ----------------
#define CIN   512
#define CO    256
#define BATCH 2
#define HH    56
#define WW    56
#define NN    (HH * WW)          // 3136
#define MTOT  (BATCH * NN)       // 6272 rows in the GEMM
#define KWIN  7
#define PAD   3
#define HALF  (CO / 2)           // 128

// ================= f32 -> bf16 conversion =================
__global__ __launch_bounds__(256) void cvt_to_bf16(const float* __restrict__ in,
                                                   __bf16* __restrict__ out, int n) {
    int i = blockIdx.x * blockDim.x + threadIdx.x;
    if (i < n) out[i] = (__bf16)in[i];
}

// ================= fused q/k/v projection GEMM (WMMA bf16) =================
// One wave computes a 16x16 tile of q, k and v (same A fragment, 3 B fragments).
// Grid: 784 blocks * 8 waves = 6272 tiles = (6272/16 row tiles) x (256/16 col tiles).
__global__ __launch_bounds__(256) void qkv_gemm_wmma(
    const __bf16* __restrict__ xb,   // (MTOT, CIN) row-major bf16
    const __bf16* __restrict__ wqb,  // (CO, CIN) row-major bf16
    const __bf16* __restrict__ wkb,
    const __bf16* __restrict__ wvb,
    float* __restrict__ q,           // (B, CO, N) f32
    float* __restrict__ k,
    float* __restrict__ v) {

    const int gwave = (blockIdx.x * blockDim.x + threadIdx.x) >> 5; // 0..6271
    const int lane  = threadIdx.x & 31;

    const int ct = gwave & 15;        // col tile (output channels)
    const int mt = gwave >> 4;        // row tile (spatial)
    const int m0 = mt << 4;
    const int c0 = ct << 4;

    const int l15   = lane & 15;
    const int hi    = lane >> 4;      // 0 or 1
    const int khalf = hi << 3;        // 0 or 8

    const __bf16* __restrict__ xrow = xb  + (size_t)(m0 + l15) * CIN;
    const __bf16* __restrict__ wqr  = wqb + (size_t)(c0 + l15) * CIN;
    const __bf16* __restrict__ wkr  = wkb + (size_t)(c0 + l15) * CIN;
    const __bf16* __restrict__ wvr  = wvb + (size_t)(c0 + l15) * CIN;

    v8f accq = {}; v8f acck = {}; v8f accv = {};

    union Frag { v16bf v; v8bf h[2]; };

    for (int ks = 0; ks < CIN; ks += 32) {
        Frag a, bq, bk, bv;
        // A: row m0+l15, K = ks+khalf..+7 and ks+khalf+16..+23  (ISA 16-bit A layout)
        a.h[0]  = *(const v8bf*)(xrow + ks + khalf);
        a.h[1]  = *(const v8bf*)(xrow + ks + khalf + 16);
        // B = W^T: column c0+l15 holds weight row (c0+l15), same K striping
        bq.h[0] = *(const v8bf*)(wqr + ks + khalf);
        bq.h[1] = *(const v8bf*)(wqr + ks + khalf + 16);
        bk.h[0] = *(const v8bf*)(wkr + ks + khalf);
        bk.h[1] = *(const v8bf*)(wkr + ks + khalf + 16);
        bv.h[0] = *(const v8bf*)(wvr + ks + khalf);
        bv.h[1] = *(const v8bf*)(wvr + ks + khalf + 16);

        accq = __builtin_amdgcn_wmma_f32_16x16x32_bf16(false, a.v, false, bq.v,
                                                       (short)0, accq, false, false);
        acck = __builtin_amdgcn_wmma_f32_16x16x32_bf16(false, a.v, false, bk.v,
                                                       (short)0, acck, false, false);
        accv = __builtin_amdgcn_wmma_f32_16x16x32_bf16(false, a.v, false, bv.v,
                                                       (short)0, accv, false, false);
    }

    // C/D layout: VGPR r -> row (r + 8*hi), col l15. For a fixed lane the 8
    // results are consecutive rows -> consecutive n -> contiguous f32 stores.
    const int m_base = m0 + (hi << 3);            // multiple of 8; NN % 8 == 0
    const int b      = m_base / NN;
    const int n      = m_base - b * NN;
    const int col    = c0 + l15;
    const size_t off = ((size_t)b * CO + col) * NN + n;

    float4* qd = (float4*)(q + off);
    float4* kd = (float4*)(k + off);
    float4* vd = (float4*)(v + off);
    qd[0] = make_float4(accq[0], accq[1], accq[2], accq[3]);
    qd[1] = make_float4(accq[4], accq[5], accq[6], accq[7]);
    kd[0] = make_float4(acck[0], acck[1], acck[2], acck[3]);
    kd[1] = make_float4(acck[4], acck[5], acck[6], acck[7]);
    vd[0] = make_float4(accv[0], accv[1], accv[2], accv[3]);
    vd[1] = make_float4(accv[4], accv[5], accv[6], accv[7]);
}

// ================= windowed softmax attention =================
// One thread per (b, c, h, w). 7x7 window fully unrolled; two passes
// (max, then exp-accumulate) to avoid a 49-register score array.
__global__ __launch_bounds__(256) void attn_window(
    const float* __restrict__ q,      // (B, CO, N)
    const float* __restrict__ k,
    const float* __restrict__ v,
    const float* __restrict__ rel_h,  // (HALF, 7)
    const float* __restrict__ rel_w,  // (HALF, 7)
    float* __restrict__ out) {        // (B, CO, N)

    const int idx = blockIdx.x * blockDim.x + threadIdx.x;   // exact: 6272*256 threads
    const int w  = idx % WW;
    const int h  = (idx / WW) % HH;
    const int c  = (idx / NN) % CO;
    // b folded into `base` via idx

    const size_t base = (size_t)(idx / NN) * NN;             // (b*CO + c)*N
    const float  qv   = q[idx];

    const bool useH = (c < HALF);
    const float* __restrict__ relp = useH ? (rel_h + c * KWIN)
                                          : (rel_w + (c - HALF) * KWIN);
    float rel7[KWIN];
#pragma unroll
    for (int i = 0; i < KWIN; ++i) rel7[i] = relp[i];

    // pass 1: max score over the 49-entry window (padding contributes q*rel)
    float smax = -3.402823466e+38f;
#pragma unroll
    for (int kh = 0; kh < KWIN; ++kh) {
#pragma unroll
        for (int kw = 0; kw < KWIN; ++kw) {
            const int hh = h + kh - PAD;
            const int ww = w + kw - PAD;
            const bool in = ((unsigned)hh < HH) & ((unsigned)ww < WW);
            const float kv = in ? k[base + hh * WW + ww] : 0.0f;
            const float rl = useH ? rel7[kh] : rel7[kw];
            const float s  = qv * (kv + rl);
            smax = fmaxf(smax, s);
        }
    }

    // pass 2: softmax numerator/denominator (v is zero at padded positions)
    float num = 0.0f, den = 0.0f;
#pragma unroll
    for (int kh = 0; kh < KWIN; ++kh) {
#pragma unroll
        for (int kw = 0; kw < KWIN; ++kw) {
            const int hh = h + kh - PAD;
            const int ww = w + kw - PAD;
            const bool in = ((unsigned)hh < HH) & ((unsigned)ww < WW);
            const int  wi = base + hh * WW + ww;
            const float kv = in ? k[wi] : 0.0f;
            const float vv = in ? v[wi] : 0.0f;
            const float rl = useH ? rel7[kh] : rel7[kw];
            const float e  = __expf(qv * (kv + rl) - smax);
            den += e;
            num += e * vv;
        }
    }
    out[idx] = num / den;
}

// ================= launch =================
extern "C" void kernel_launch(void* const* d_in, const int* in_sizes, int n_in,
                              void* d_out, int out_size, void* d_ws, size_t ws_size,
                              hipStream_t stream) {
    const float* x     = (const float*)d_in[0];   // (2, 3136, 512)
    const float* w_q   = (const float*)d_in[1];   // (256, 512)
    const float* w_k   = (const float*)d_in[2];
    const float* w_v   = (const float*)d_in[3];
    const float* rel_h = (const float*)d_in[4];   // (128, 7)
    const float* rel_w = (const float*)d_in[5];   // (128, 7)
    float* out = (float*)d_out;                   // (2, 256*56, 56)

    // ---- workspace carve-up (all offsets 256B aligned) ----
    char* ws = (char*)d_ws;
    const size_t XB_ELEMS = (size_t)MTOT * CIN;       // 3,211,264
    const size_t W_ELEMS  = (size_t)CO * CIN;         // 131,072
    const size_t QKV_ELEM = (size_t)BATCH * CO * NN;  // 1,605,632

    __bf16* xb  = (__bf16*)(ws);
    __bf16* wqb = (__bf16*)(ws + XB_ELEMS * 2);
    __bf16* wkb = (__bf16*)(ws + XB_ELEMS * 2 + W_ELEMS * 2);
    __bf16* wvb = (__bf16*)(ws + XB_ELEMS * 2 + W_ELEMS * 4);
    float*  qf  = (float*) (ws + XB_ELEMS * 2 + W_ELEMS * 6);
    float*  kf  = qf + QKV_ELEM;
    float*  vf  = kf + QKV_ELEM;

    // ---- 1) convert x and weights to bf16 ----
    cvt_to_bf16<<<(int)((XB_ELEMS + 255) / 256), 256, 0, stream>>>(x, xb, (int)XB_ELEMS);
    cvt_to_bf16<<<(int)((W_ELEMS + 255) / 256), 256, 0, stream>>>(w_q, wqb, (int)W_ELEMS);
    cvt_to_bf16<<<(int)((W_ELEMS + 255) / 256), 256, 0, stream>>>(w_k, wkb, (int)W_ELEMS);
    cvt_to_bf16<<<(int)((W_ELEMS + 255) / 256), 256, 0, stream>>>(w_v, wvb, (int)W_ELEMS);

    // ---- 2) fused q/k/v projection: 6272 16x16 tiles, 8 waves/block ----
    qkv_gemm_wmma<<<MTOT * (CO / 16) / 16 / 8, 256, 0, stream>>>(xb, wqb, wkb, wvb,
                                                                 qf, kf, vf);

    // ---- 3) windowed softmax attention ----
    attn_window<<<(int)(QKV_ELEM / 256), 256, 0, stream>>>(qf, kf, vf, rel_h, rel_w, out);
}
